// LpAlignEntropyLoss_6253472383015
// MI455X (gfx1250) — compile-verified, matrix operand order checked by reading, not run on hardware
//
#include <hip/hip_runtime.h>
#include <hip/hip_bf16.h>

// ---------------------------------------------------------------------------
// LpAlignEntropyLoss on MI455X (gfx1250)
//
// loss = (ALPHA*pos + (1-ALPHA)*neg)/3 over pairs (z1,z2),(z1,z3),(z2,z3)
//   pos term : mean_a ||zi_a - zj_a||
//   neg term : mean_a [ log( mean_b exp(-dist(a,b)/TAU) ) ]
//   dist(a,b) = sqrt(max(|zi_a|^2 + |zj_b|^2 - 2 zi_a.zj_b, 0))
//
// Strategy: fused tile GEMM (bf16 split-3 WMMA, fp32 accumulate) + distance
// epilogue + per-row exp-sum accumulation. Never materializes the 8192x8192
// Gram matrices (805 MB) -> everything stays in registers/LDS/L2.
// Workspace layout (floats): [0,24576) rowsum[3][8192]; [24576] posAcc;
// [24577] negAcc.  Needs >= 98312 bytes of d_ws.
// ---------------------------------------------------------------------------

#define B_ROWS 8192
#define DIMK   128
#define NPAIR  3
#define TILE   128
#define PITCH  132   // bf16 elems per LDS row: 128 + 4 pad (breaks 64-bank stride)

typedef __attribute__((ext_vector_type(16))) __bf16 v16bf;
typedef __attribute__((ext_vector_type(8)))  float  v8f;

union FragU { v16bf v; unsigned short u[16]; };

__device__ __forceinline__ unsigned short f2bf(float x) {
    unsigned int u = __float_as_uint(x);
    u += 0x7FFFu + ((u >> 16) & 1u);          // round-to-nearest-even
    return (unsigned short)(u >> 16);
}
__device__ __forceinline__ float bf2f(unsigned short h) {
    return __uint_as_float(((unsigned int)h) << 16);
}

// A/B fragment gather from LDS (bf16 16x32 operand layout, ISA 7.12.2):
// lane l: row/col = l&15, half h = l>>4; VGPR v holds K pair at
// kbase = (v<4 ? 2v : 16+2(v-4)) + 8h  within the 32-wide K chunk.
__device__ __forceinline__ void load_frag(const unsigned short* s, int rcBase,
                                          int kc, int lane, FragU& f) {
    int rc = rcBase + (lane & 15);
    int h  = lane >> 4;
    const unsigned short* p = s + rc * PITCH + kc * 32 + h * 8;
#pragma unroll
    for (int v = 0; v < 8; ++v) {
        int kb = (v < 4) ? (v * 2) : (16 + (v - 4) * 2);
        f.u[2 * v]     = p[kb];
        f.u[2 * v + 1] = p[kb + 1];
    }
}

// ---------------------------------------------------------------------------
__global__ void zero_ws_kernel(float* ws, int n) {
    int i = blockIdx.x * blockDim.x + threadIdx.x;
    if (i < n) ws[i] = 0.0f;
}

// ---------------------------------------------------------------------------
// pos term: one wave per (pair,row).  grid = 3*8192/8 blocks of 256.
__global__ void pos_kernel(const float* __restrict__ z1,
                           const float* __restrict__ z2,
                           const float* __restrict__ z3,
                           float* posAcc) {
    __shared__ float wsum[8];
    int lane = threadIdx.x & 31;
    int wid  = threadIdx.x >> 5;
    int idx  = blockIdx.x * 8 + wid;          // [0, 3*8192)
    int p = idx >> 13;
    int a = idx & (B_ROWS - 1);
    const float* vi = (p < 2) ? z1 : z2;
    const float* vj = (p == 0) ? z2 : z3;
    const float4* ra = (const float4*)(vi + (size_t)a * DIMK);
    const float4* rb = (const float4*)(vj + (size_t)a * DIMK);
    float4 x = ra[lane], y = rb[lane];
    float d0 = x.x - y.x, d1 = x.y - y.y, d2 = x.z - y.z, d3 = x.w - y.w;
    float s = d0 * d0 + d1 * d1 + d2 * d2 + d3 * d3;
    s += __shfl_xor(s, 16); s += __shfl_xor(s, 8); s += __shfl_xor(s, 4);
    s += __shfl_xor(s, 2);  s += __shfl_xor(s, 1);
    if (lane == 0) wsum[wid] = sqrtf(s);
    __syncthreads();
    if (threadIdx.x == 0) {
        float tot = 0.f;
#pragma unroll
        for (int k = 0; k < 8; ++k) tot += wsum[k];
        atomicAdd(posAcc, tot);
    }
}

// ---------------------------------------------------------------------------
// Fused 128x128 tile: bf16 split-3 WMMA GEMM + exp(-dist) + row sums.
// grid = 3 * 64 * 64 blocks of 256 threads (8 wave32).
__global__ __launch_bounds__(256) void tile_kernel(const float* __restrict__ z1,
                                                   const float* __restrict__ z2,
                                                   const float* __restrict__ z3,
                                                   float* __restrict__ rowsum) {
    __shared__ unsigned short sAhi[TILE * PITCH];
    __shared__ unsigned short sAlo[TILE * PITCH];
    __shared__ unsigned short sBhi[TILE * PITCH];
    __shared__ unsigned short sBlo[TILE * PITCH];
    __shared__ float s_a2[TILE];
    __shared__ float s_b2[TILE];

    int bx  = blockIdx.x;
    int p   = bx >> 12;                 // 4096 tiles per pair
    int rem = bx & 4095;
    int m0  = (rem >> 6) << 7;          // row tile base
    int n0  = (rem & 63) << 7;          // col tile base

    const float* A  = (p == 2) ? z2 : z1;   // rows of the distance matrix
    const float* Bm = (p == 0) ? z2 : z3;   // cols of the distance matrix

    int t = threadIdx.x;
    if (t < 128) s_a2[t] = 0.f; else s_b2[t - 128] = 0.f;
    __syncthreads();

    // --- load both 128x128 f32 tiles, split fp32 -> (hi,lo) bf16, row norms
    {
        int r  = t >> 1;                // 0..127 : tile row
        int hh = (t & 1) * 64;          // which 64-col half of the row
        const float4* ga = (const float4*)(A  + (size_t)(m0 + r) * DIMK + hh);
        const float4* gb = (const float4*)(Bm + (size_t)(n0 + r) * DIMK + hh);
        float sa = 0.f, sb = 0.f;
#pragma unroll
        for (int it = 0; it < 16; ++it) {
            float4 va = ga[it];
            float4 vb = gb[it];
            int c = hh + it * 4;
            float av[4] = {va.x, va.y, va.z, va.w};
            float bv[4] = {vb.x, vb.y, vb.z, vb.w};
#pragma unroll
            for (int q = 0; q < 4; ++q) {
                float xa = av[q];
                unsigned short ha = f2bf(xa);
                sAhi[r * PITCH + c + q] = ha;
                sAlo[r * PITCH + c + q] = f2bf(xa - bf2f(ha));
                sa += xa * xa;
                float xb = bv[q];
                unsigned short hb = f2bf(xb);
                sBhi[r * PITCH + c + q] = hb;
                sBlo[r * PITCH + c + q] = f2bf(xb - bf2f(hb));
                sb += xb * xb;
            }
        }
        atomicAdd(&s_a2[r], sa);
        atomicAdd(&s_b2[r], sb);
    }
    __syncthreads();

    // --- per-wave 32x64 region: 2 M-subtiles x 4 N-subtiles of 16x16
    int lane  = threadIdx.x & 31;
    int w     = threadIdx.x >> 5;
    int Rbase = (w >> 1) * 32;
    int Cbase = (w & 1) * 64;

    v8f acc[2][4] = {};

    for (int kc = 0; kc < 4; ++kc) {                 // K chunks of 32
        FragU ahi[2], alo[2], bhi[4], blo[4];
#pragma unroll
        for (int mi = 0; mi < 2; ++mi) {
            load_frag(sAhi, Rbase + mi * 16, kc, lane, ahi[mi]);
            load_frag(sAlo, Rbase + mi * 16, kc, lane, alo[mi]);
        }
#pragma unroll
        for (int ni = 0; ni < 4; ++ni) {
            load_frag(sBhi, Cbase + ni * 16, kc, lane, bhi[ni]);
            load_frag(sBlo, Cbase + ni * 16, kc, lane, blo[ni]);
        }
#pragma unroll
        for (int mi = 0; mi < 2; ++mi)
#pragma unroll
            for (int ni = 0; ni < 4; ++ni) {
                // split-3: hi*hi + hi*lo + lo*hi  (fp32 accumulate)
                acc[mi][ni] = __builtin_amdgcn_wmma_f32_16x16x32_bf16(
                    false, ahi[mi].v, false, bhi[ni].v, (short)0, acc[mi][ni], false, false);
                acc[mi][ni] = __builtin_amdgcn_wmma_f32_16x16x32_bf16(
                    false, ahi[mi].v, false, blo[ni].v, (short)0, acc[mi][ni], false, false);
                acc[mi][ni] = __builtin_amdgcn_wmma_f32_16x16x32_bf16(
                    false, alo[mi].v, false, bhi[ni].v, (short)0, acc[mi][ni], false, false);
            }
    }

    // --- epilogue: dist -> exp -> per-row sums (C/D layout: VGPR r, lane l:
    //     M = r + 8*(l>>4), N = l&15 within each 16x16 subtile)
    int half = lane >> 4;
    int lcol = lane & 15;
    float rowpart[2][8];
#pragma unroll
    for (int mi = 0; mi < 2; ++mi)
#pragma unroll
        for (int r = 0; r < 8; ++r) rowpart[mi][r] = 0.f;

#pragma unroll
    for (int ni = 0; ni < 4; ++ni) {
        float bn = s_b2[Cbase + ni * 16 + lcol];
#pragma unroll
        for (int mi = 0; mi < 2; ++mi) {
#pragma unroll
            for (int r = 0; r < 8; ++r) {
                float am = s_a2[Rbase + mi * 16 + r + 8 * half];
                float g  = acc[mi][ni][r];
                float sq = fmaxf(am + bn - 2.0f * g, 0.0f);
                float e  = __expf(-sqrtf(sq));      // TAU == 1.0
                // sum the 16 columns held by this half-wave
                e += __shfl_xor(e, 1); e += __shfl_xor(e, 2);
                e += __shfl_xor(e, 4); e += __shfl_xor(e, 8);
                rowpart[mi][r] += e;
            }
        }
    }

    if (lcol == 0) {
        float* rs = rowsum + (size_t)p * B_ROWS;
#pragma unroll
        for (int mi = 0; mi < 2; ++mi)
#pragma unroll
            for (int r = 0; r < 8; ++r) {
                int grow = m0 + Rbase + mi * 16 + r + 8 * half;
                atomicAdd(&rs[grow], rowpart[mi][r]);
            }
    }
}

// ---------------------------------------------------------------------------
// neg term: logmeanexp per row, block-reduced before the single-address atomic.
__global__ void finalize_kernel(const float* __restrict__ rowsum, float* negAcc) {
    __shared__ float wsum[8];
    int i    = blockIdx.x * blockDim.x + threadIdx.x;
    int lane = threadIdx.x & 31;
    int wid  = threadIdx.x >> 5;
    float v = 0.f;
    if (i < NPAIR * B_ROWS) v = logf(rowsum[i]) - logf((float)B_ROWS);
    v += __shfl_xor(v, 16); v += __shfl_xor(v, 8); v += __shfl_xor(v, 4);
    v += __shfl_xor(v, 2);  v += __shfl_xor(v, 1);
    if (lane == 0) wsum[wid] = v;
    __syncthreads();
    if (threadIdx.x == 0) {
        float tot = 0.f;
#pragma unroll
        for (int k = 0; k < 8; ++k) tot += wsum[k];
        atomicAdd(negAcc, tot);
    }
}

__global__ void writeout_kernel(const float* posAcc, const float* negAcc, float* out) {
    if (threadIdx.x == 0 && blockIdx.x == 0) {
        float pos = posAcc[0] * (1.0f / (float)B_ROWS);   // sum over pairs of means
        float neg = negAcc[0] * (1.0f / (float)B_ROWS);
        out[0] = (0.5f * pos + 0.5f * neg) * (1.0f / 3.0f);  // ALPHA = 0.5, 3 pairs
    }
}

// ---------------------------------------------------------------------------
extern "C" void kernel_launch(void* const* d_in, const int* in_sizes, int n_in,
                              void* d_out, int out_size, void* d_ws, size_t ws_size,
                              hipStream_t stream) {
    (void)in_sizes; (void)n_in; (void)out_size; (void)ws_size;
    const float* z1 = (const float*)d_in[0];
    const float* z2 = (const float*)d_in[1];
    const float* z3 = (const float*)d_in[2];

    float* ws      = (float*)d_ws;
    float* rowsum  = ws;                       // [3][8192]
    float* posAcc  = ws + NPAIR * B_ROWS;      // [1]
    float* negAcc  = ws + NPAIR * B_ROWS + 1;  // [1]
    const int nWs  = NPAIR * B_ROWS + 2;

    zero_ws_kernel<<<(nWs + 255) / 256, 256, 0, stream>>>(ws, nWs);

    pos_kernel<<<(NPAIR * B_ROWS) / 8, 256, 0, stream>>>(z1, z2, z3, posAcc);

    tile_kernel<<<NPAIR * (B_ROWS / TILE) * (B_ROWS / TILE), 256, 0, stream>>>(
        z1, z2, z3, rowsum);

    finalize_kernel<<<(NPAIR * B_ROWS) / 256, 256, 0, stream>>>(rowsum, negAcc);

    writeout_kernel<<<1, 32, 0, stream>>>(posAcc, negAcc, (float*)d_out);
}
